// Model_69801808494712
// MI455X (gfx1250) — compile-verified, matrix operand order checked by reading, not run on hardware
//
#include <hip/hip_runtime.h>
#include <hip/hip_bf16.h>
#include <cstddef>

// ---------------------------------------------------------------------------
// Sizes from the reference
// ---------------------------------------------------------------------------
#define BB   8
#define LL   48000
#define VV   28
#define QQ   256
#define CC   256
#define KK1  10
#define KK2  5
#define DSZ  50          // K1*K2
#define TT   960         // L/DS
#define TT1  4800        // L/K1
#define NROWS (BB*TT1)   // 38400 GEMM rows
#define NCOLS (QQ*KK1)   // 2560 GEMM cols
#define NTILES (NCOLS/16) // 160 N-tiles
#define KSTEPS (CC/32)    // 8 K-steps of 32 (f16 WMMA)

typedef __attribute__((ext_vector_type(16))) _Float16 v16h;
typedef __attribute__((ext_vector_type(8)))  float    v8f;

// Workspace layout (bytes)
#define ACC_OFF    0        // float[2]: {rec_sum, kl_sum}
#define VIDX_OFF   256      // int[7680]
#define BFRAG_OFF  30976    // _Float16[160*8*32*16]  (1.31 MB), 32B aligned

// ---------------------------------------------------------------------------
__global__ void init_kernel(float* acc) {
    if (threadIdx.x < 2) acc[threadIdx.x] = 0.0f;
}

// ---------------------------------------------------------------------------
// Pre-swizzle Wd2 (C,Q,K1) f32 -> f16 WMMA B-fragments.
// B fragment (32x16 f16, wave32), assumed transpose-symmetric to A layout:
//  lane<16 : n=lane,    elems 0..7 -> K=0..7,  elems 8..15 -> K=16..23
//  lane>=16: n=lane-16, elems 0..7 -> K=8..15, elems 8..15 -> K=24..31
// ---------------------------------------------------------------------------
__global__ void bfrag_prep_kernel(const float* __restrict__ Wd2,
                                  _Float16* __restrict__ bfrag) {
    int g = blockIdx.x * blockDim.x + threadIdx.x;   // (nt*8+kk)*32+lane
    if (g >= NTILES * KSTEPS * 32) return;
    int lane = g & 31;
    int kk   = (g >> 5) & 7;
    int nt   = g >> 8;
    int hi   = lane >> 4;
    int n    = lane & 15;
    int col  = nt * 16 + n;                          // q*10+jj
    #pragma unroll
    for (int s = 0; s < 16; ++s) {
        int klocal = s + ((s >= 8) ? 8 : 0) + hi * 8;
        int c = kk * 32 + klocal;
        float v = Wd2[(size_t)c * NCOLS + col];      // (C,Q,K1) flat: c*2560+col
        bfrag[(size_t)g * 16 + s] = (_Float16)v;
    }
}

// ---------------------------------------------------------------------------
// Fused encoder: h = clip(conv1,0,6) -> logits (28) -> +gumbel -> argmax.
// One wave32 per code position t; 8 waves per block.
// ---------------------------------------------------------------------------
__global__ void enc_argmax_kernel(const float* __restrict__ x,
                                  const float* __restrict__ W1,
                                  const float* __restrict__ b1,
                                  const float* __restrict__ W2,
                                  const float* __restrict__ b2,
                                  const float* __restrict__ gumbel,
                                  int* __restrict__ vidx) {
    __shared__ float xbuf[8][52];
    __shared__ float hbuf[8][1280];
    int wv = threadIdx.x >> 5;
    int lane = threadIdx.x & 31;
    int u = blockIdx.x * 8 + wv;            // b*960+t, grid sized exactly
    int b = u / TT, t = u % TT;
    const float* xp = x + (size_t)b * LL + t * DSZ;
    if (lane < DSZ)      xbuf[wv][lane]      = xp[lane];
    if (lane + 32 < DSZ) xbuf[wv][lane + 32] = xp[lane + 32];
    __syncthreads();
    // h[c*5+j], 40 elements per lane
    for (int s = 0; s < 40; ++s) {
        int i = lane + 32 * s;              // c*5+j
        int c = i / 5, j = i % 5;
        float acc = b1[c];
        const float* w  = W1 + c * KK1;
        const float* xx = &xbuf[wv][j * KK1];
        #pragma unroll
        for (int k = 0; k < KK1; ++k) acc = fmaf(xx[k], w[k], acc);
        hbuf[wv][i] = fminf(fmaxf(acc, 0.0f), 6.0f);
    }
    __syncthreads();
    float best = -3.4e38f; int bi = lane;
    if (lane < VV) {
        const float* w2 = W2 + (size_t)lane * 1280;   // (V,C,K2): v*1280 + c*5+j
        float acc = b2[lane];
        for (int i = 0; i < 1280; ++i) acc = fmaf(hbuf[wv][i], w2[i], acc);
        best = acc + gumbel[(size_t)u * VV + lane];
    }
    #pragma unroll
    for (int off = 16; off > 0; off >>= 1) {
        float ob = __shfl_down(best, off);
        int   oi = __shfl_down(bi,   off);
        if (ob > best) { best = ob; bi = oi; }
    }
    if (lane == 0) vidx[u] = bi;
}

// ---------------------------------------------------------------------------
// KL: one-hot z -> per active t: log(1/(ngrams[p0,p1,v]+1e-10)+1e-10)
// ---------------------------------------------------------------------------
__global__ void kl_kernel(const int* __restrict__ vidx,
                          const int* __restrict__ x_sl,
                          const float* __restrict__ ngrams,
                          float* __restrict__ acc) {
    int u = blockIdx.x * blockDim.x + threadIdx.x;
    if (u >= BB * TT) return;
    int b = u / TT, t = u % TT;
    int zsl = x_sl[b] / DSZ;
    if (t < zsl) {
        int v  = vidx[u];
        int p0 = (t >= 2) ? vidx[u - 2] : VV;
        int p1 = (t >= 1) ? vidx[u - 1] : VV;
        float prior = ngrams[((size_t)p0 * (VV + 1) + p1) * VV + v];
        float term = logf(1.0f / (prior + 1e-10f) + 1e-10f);
        atomicAdd(acc + 1, term);
    }
}

// ---------------------------------------------------------------------------
// Decoder fused: build A tile (16 rows x 256 f16) from one-hot gather of Wd1,
// WMMA GEMM vs pre-swizzled Wd2 fragments into a 16x2560 f32 LDS slab,
// then log-softmax over Q=256 per (row,jj) + mu-law-target NLL accumulation.
// Block: 512 threads = 16 waves; rows-per-block = 16 (4800%16==0 -> b uniform).
// Dynamic LDS: 16*2560*4 + 16*256*2 + 256*4 = 173056 B (< 320 KB/WGP).
// ---------------------------------------------------------------------------
__global__ void dec_gemm_softmax_kernel(const float* __restrict__ x,
                                        const int* __restrict__ x_sl,
                                        const int* __restrict__ vidx,
                                        const float* __restrict__ Wd1,
                                        const float* __restrict__ bd1,
                                        const _Float16* __restrict__ bfrag,
                                        const float* __restrict__ bd2,
                                        float* __restrict__ acc) {
    extern __shared__ char smem[];
    float*    outb = (float*)smem;                                  // [16][2560]
    _Float16* As   = (_Float16*)(smem + 16 * NCOLS * 4);            // [16][256]
    float*    bd2s = (float*)(smem + 16 * NCOLS * 4 + 16 * CC * 2); // [256]

    const int tid  = threadIdx.x;
    const int lane = tid & 31;
    const int w    = tid >> 5;
    const int r0   = blockIdx.x * 16;        // row = b*4800 + t1
    const int b    = r0 / TT1;
    const int zsl  = x_sl[b] / DSZ;

    if (tid < QQ) bd2s[tid] = bd2[tid];

    // Phase 1: synthesize A tile: r1 = clip((t<zsl ? Wd1[v,c,j] : 0)+bd1[c],0,6)
    for (int e = tid; e < 16 * CC; e += blockDim.x) {
        int r = e >> 8, c = e & 255;
        int t1 = (r0 + r) % TT1;
        int t = t1 / KK2, j = t1 % KK2;
        float val = bd1[c];
        if (t < zsl) {
            int v = vidx[b * TT + t];
            val += Wd1[((size_t)v * CC + c) * KK2 + j];
        }
        val = fminf(fmaxf(val, 0.0f), 6.0f);
        As[r * CC + c] = (_Float16)val;
    }
    __syncthreads();

    // Phase 2: A fragments for all 8 K-steps (ISA 7.12.2, 16-bit A 16x32)
    v16h afrag[KSTEPS];
    {
        int m  = lane & 15;
        int hi = (lane >> 4) & 1;
        #pragma unroll
        for (int kk = 0; kk < KSTEPS; ++kk) {
            const _Float16* p0 = &As[m * CC + kk * 32 + hi * 8];
            const _Float16* p1 = &As[m * CC + kk * 32 + 16 + hi * 8];
            #pragma unroll
            for (int s = 0; s < 8; ++s) {
                afrag[kk][s]     = p0[s];
                afrag[kk][s + 8] = p1[s];
            }
        }
    }

    // Phase 3: each wave covers 10 N-tiles (nt = w, w+16, ..., w+144)
    for (int it = 0; it < 10; ++it) {
        int nt = w + 16 * it;
        v8f c8 = {};
        #pragma unroll
        for (int kk = 0; kk < KSTEPS; ++kk) {
            v16h bf = *reinterpret_cast<const v16h*>(
                bfrag + (((size_t)nt * KSTEPS + kk) * 32 + lane) * 16);
            c8 = __builtin_amdgcn_wmma_f32_16x16x32_f16(
                false, afrag[kk], false, bf, (short)0, c8, false, false);
        }
        // D layout: VGPR p -> M=p (lanes 0-15) / M=8+p (lanes 16-31); N=lane%16
        int col = nt * 16 + (lane & 15);
        int rb  = (lane < 16) ? 0 : 8;
        #pragma unroll
        for (int p = 0; p < 8; ++p) outb[(rb + p) * NCOLS + col] = c8[p];
    }
    __syncthreads();

    // Phase 4: wave w owns row w: log-softmax over q (cols q*10+jj) + NLL
    float lsum = 0.0f;
    {
        const int r  = w;
        const int t1 = (r0 + r) % TT1;
        const int xsl = x_sl[b];
        const float* row = &outb[r * NCOLS];
        const float inv_log1p_mu = 1.0f / log1pf(255.0f);
        for (int jj = 0; jj < KK1; ++jj) {
            float vq[8];
            float mx = -3.4e38f;
            #pragma unroll
            for (int s = 0; s < 8; ++s) {
                int q = lane + 32 * s;
                vq[s] = row[q * KK1 + jj] + bd2s[q];
                mx = fmaxf(mx, vq[s]);
            }
            #pragma unroll
            for (int off = 16; off > 0; off >>= 1)
                mx = fmaxf(mx, __shfl_xor(mx, off));
            float e = 0.0f;
            #pragma unroll
            for (int s = 0; s < 8; ++s) e += expf(vq[s] - mx);
            #pragma unroll
            for (int off = 16; off > 0; off >>= 1) e += __shfl_xor(e, off);
            if (lane == 0) {
                int l = t1 * KK1 + jj;
                if (l < xsl) {
                    float xv = x[(size_t)b * LL + l];
                    float y  = ((xv >= 0.0f) ? 1.0f : -1.0f) *
                               log1pf(255.0f * fabsf(xv)) * inv_log1p_mu;
                    int tc = (int)floorf((y + 1.0f) * 0.5f * (float)QQ);
                    tc = tc < 0 ? 0 : (tc > QQ - 1 ? QQ - 1 : tc);
                    float r2t = row[tc * KK1 + jj] + bd2s[tc];
                    lsum += -(r2t - mx - logf(e));   // -log_softmax[target]
                }
            }
        }
    }
    if (lane == 0) atomicAdd(acc, lsum);
}

// ---------------------------------------------------------------------------
__global__ void finalize_kernel(const float* __restrict__ acc,
                                const int* __restrict__ x_sl,
                                float* __restrict__ out) {
    if (blockIdx.x == 0 && threadIdx.x == 0) {
        int xs = 0, zs = 0;
        for (int b = 0; b < BB; ++b) { xs += x_sl[b]; zs += x_sl[b] / DSZ; }
        out[0] = acc[0] / (float)xs + acc[1] / (float)zs;
    }
}

// ---------------------------------------------------------------------------
extern "C" void kernel_launch(void* const* d_in, const int* in_sizes, int n_in,
                              void* d_out, int out_size, void* d_ws, size_t ws_size,
                              hipStream_t stream) {
    const float* x      = (const float*)d_in[0];
    const int*   x_sl   = (const int*)d_in[1];
    const float* ngrams = (const float*)d_in[4];
    const float* gumbel = (const float*)d_in[5];
    const float* W1     = (const float*)d_in[6];
    const float* b1     = (const float*)d_in[7];
    const float* W2     = (const float*)d_in[8];
    const float* b2     = (const float*)d_in[9];
    const float* Wd1    = (const float*)d_in[10];
    const float* bd1    = (const float*)d_in[11];
    const float* Wd2    = (const float*)d_in[12];
    const float* bd2    = (const float*)d_in[13];

    char*     ws    = (char*)d_ws;
    float*    acc   = (float*)(ws + ACC_OFF);
    int*      vidx  = (int*)(ws + VIDX_OFF);
    _Float16* bfrag = (_Float16*)(ws + BFRAG_OFF);

    const size_t smem = (size_t)16 * NCOLS * 4 + (size_t)16 * CC * 2 + QQ * 4;
    hipFuncSetAttribute((const void*)dec_gemm_softmax_kernel,
                        hipFuncAttributeMaxDynamicSharedMemorySize, (int)smem);

    init_kernel<<<1, 32, 0, stream>>>(acc);
    bfrag_prep_kernel<<<(NTILES * KSTEPS * 32 + 255) / 256, 256, 0, stream>>>(Wd2, bfrag);
    enc_argmax_kernel<<<(BB * TT) / 8, 256, 0, stream>>>(x, W1, b1, W2, b2, gumbel, vidx);
    kl_kernel<<<(BB * TT + 255) / 256, 256, 0, stream>>>(vidx, x_sl, ngrams, acc);
    dec_gemm_softmax_kernel<<<NROWS / 16, 512, smem, stream>>>(
        x, x_sl, vidx, Wd1, bd1, bfrag, bd2, acc);
    finalize_kernel<<<1, 32, 0, stream>>>(acc, x_sl, (float*)d_out);
}
